// SOHMM_76175539962077
// MI455X (gfx1250) — compile-verified, matrix operand order checked by reading, not run on hardware
//
#include <hip/hip_runtime.h>
#include <hip/hip_bf16.h>
#include <math.h>

// ---------------------------------------------------------------------------
// SOHMM forward for MI455X (gfx1250, wave32, WMMA).
//   H=128, V=32000, B=64, S=128
//   Per step t: y_new[i,j,b] = log( sum_k alpha[i,j,k]*exp(y[j,k,b]-ymax[b]) + eps )
//                              + ymax[b] + beta[j, ids[b,t]]
// Core GEMM per (step, j): [128x64] = [128x128] x [128x64] via
// v_wmma_f32_16x16x32_f16 (f16 inputs in [0,1], f32 accumulate).
// ---------------------------------------------------------------------------

#define H 128
#define V 32000
#define B 64
#define S 128
#define SLICE (H * H * B) // 1,048,576 floats per time slice

typedef __attribute__((ext_vector_type(16))) _Float16 v16h;
typedef __attribute__((ext_vector_type(8)))  _Float16 v8h;
typedef __attribute__((ext_vector_type(8)))  float    v8f;
typedef __attribute__((ext_vector_type(4)))  float    v4f;

#define ENC_NEG_INF 0x007FFFFFu // enc(-inf)

// Monotonic float<->uint mapping so we can use integer atomicMax for float max.
__device__ __forceinline__ unsigned encf(float f) {
    unsigned b = __float_as_uint(f);
    return (b & 0x80000000u) ? ~b : (b | 0x80000000u);
}
__device__ __forceinline__ float decf(unsigned u) {
    return __uint_as_float((u & 0x80000000u) ? (u ^ 0x80000000u) : ~u);
}

// ---------------- gamma softmax stats: gstats = {gmax, gsum} ----------------
__global__ __launch_bounds__(256) void sohmm_gamma_stats(const float* __restrict__ g,
                                                         float* __restrict__ gstats) {
    __shared__ float red[256];
    const int tid = threadIdx.x;
    float m = -INFINITY;
    for (int i = tid; i < H * H; i += 256) m = fmaxf(m, g[i]);
    red[tid] = m; __syncthreads();
    for (int s = 128; s > 0; s >>= 1) {
        if (tid < s) red[tid] = fmaxf(red[tid], red[tid + s]);
        __syncthreads();
    }
    const float gmax = red[0]; __syncthreads();
    float sum = 0.f;
    for (int i = tid; i < H * H; i += 256) sum += __expf(g[i] - gmax);
    red[tid] = sum; __syncthreads();
    for (int s = 128; s > 0; s >>= 1) {
        if (tid < s) red[tid] += red[tid + s];
        __syncthreads();
    }
    if (tid == 0) { gstats[0] = gmax; gstats[1] = red[0]; }
}

// ---------------- init rotating max buffers (3 x 64, encoded) ----------------
__global__ __launch_bounds__(256) void sohmm_init(unsigned* __restrict__ buf) {
    if (threadIdx.x < 3 * B) buf[threadIdx.x] = ENC_NEG_INF;
}

// ---------------- y0: out[0][i][j][b] = beta[j, ids[b, S-1]] ----------------
__global__ __launch_bounds__(256) void sohmm_y0(const int* __restrict__ ids,
                                                const float* __restrict__ beta,
                                                float* __restrict__ out,
                                                unsigned* __restrict__ buf) {
    const int j = blockIdx.x;
    const int tid = threadIdx.x;
    __shared__ float ipv[B];
    if (tid < B) {
        const int id = ids[tid * S + (S - 1)];
        const float v = (id < 0) ? 0.f : beta[(size_t)j * V + id];
        ipv[tid] = v;
        atomicMax(&buf[0 * B + tid], encf(v));   // max of y0 slice (indep of i)
        buf[1 * B + tid] = ENC_NEG_INF;          // init for step 1 (benign race)
    }
    __syncthreads();
    const int b = tid & 63;
    const float v = ipv[b];
    for (int it = 0; it < 32; ++it) {
        const int i = it * 4 + (tid >> 6);
        out[((size_t)i * H + j) * B + b] = v;
    }
}

// ---------------- one recurrence step (grid: 128 blocks, one per j) ----------
__global__ __launch_bounds__(256) void sohmm_step(const int s,
                                                  const int* __restrict__ ids,
                                                  const float* __restrict__ alpha,
                                                  const float* __restrict__ beta,
                                                  float* __restrict__ out,
                                                  unsigned* __restrict__ buf) {
    const int j   = blockIdx.x;
    const int tid = threadIdx.x;

    __shared__ _Float16 yexp[B * 136];  // [b][k], stride 136 halves (conflict-free b128 reads)
    __shared__ float    ymaxs[B];
    __shared__ float    ipv[B];
    __shared__ unsigned smax[B];

    const unsigned* bufPrev = buf + ((s - 1) % 3) * B;
    unsigned*       bufCur  = buf + (s % 3) * B;
    unsigned*       bufNext = buf + ((s + 1) % 3) * B;

    if (tid < B) {
        ymaxs[tid] = decf(bufPrev[tid]);
        smax[tid]  = ENC_NEG_INF;
        bufNext[tid] = ENC_NEG_INF;              // pre-init for step s+1 (benign race)
        const int t  = (S - 1) - s;              // emission time index
        const int id = ids[tid * S + t];
        ipv[tid] = (id < 0) ? 0.f : beta[(size_t)j * V + id];
    }
    __syncthreads();

    // y_exp[j,k,b] = exp(y_prev[j,k,b] - ymax[b]) -> LDS transposed [b][k] in f16
    const float* yprev = out + (size_t)(s - 1) * SLICE + (size_t)j * H * B;
    for (int e = tid; e < H * B; e += 256) {     // e = k*64 + b, coalesced read
        const int b = e & 63;
        const int k = e >> 6;
        yexp[b * 136 + k] = (_Float16)__expf(yprev[e] - ymaxs[b]);
    }
    __syncthreads();

    // 8 waves; wave w owns M-tile w (rows i = w*16 .. w*16+15), all 4 N-tiles.
    const int wv   = tid >> 5;
    const int lane = tid & 31;
    const int m    = lane & 15;
    const int half = lane >> 4;

    const float* arow = alpha + ((size_t)(wv * 16 + m) * H + j) * H; // A_j row, contiguous K

    v8f c[4];
    #pragma unroll
    for (int nt = 0; nt < 4; ++nt) c[nt] = (v8f)(0.f);

    #pragma unroll
    for (int kt = 0; kt < 4; ++kt) {
        const int k0  = kt * 32;
        const int klo = k0 + half * 8;
        const int khi = klo + 16;
        // A fragment: f32 load + convert (alpha is L2-resident; conversion is cheap VALU)
        const v4f f0 = *(const v4f*)(arow + klo);
        const v4f f1 = *(const v4f*)(arow + klo + 4);
        const v4f f2 = *(const v4f*)(arow + khi);
        const v4f f3 = *(const v4f*)(arow + khi + 4);
        v16h a;
        #pragma unroll
        for (int e = 0; e < 4; ++e) {
            a[e]      = (_Float16)f0[e];
            a[e + 4]  = (_Float16)f1[e];
            a[e + 8]  = (_Float16)f2[e];
            a[e + 12] = (_Float16)f3[e];
        }
        #pragma unroll
        for (int nt = 0; nt < 4; ++nt) {
            const int b = nt * 16 + m;           // column of B fragment
            const v8h blo = *(const v8h*)(&yexp[b * 136 + klo]);
            const v8h bhi = *(const v8h*)(&yexp[b * 136 + khi]);
            v16h bb;
            #pragma unroll
            for (int e = 0; e < 8; ++e) { bb[e] = blo[e]; bb[e + 8] = bhi[e]; }
            c[nt] = __builtin_amdgcn_wmma_f32_16x16x32_f16(
                false, a, false, bb, (short)0, c[nt], false, false);
        }
    }

    // Epilogue: y_new = log(acc + eps) + ymax_prev + ip ; feed next-step max.
    float* outS = out + (size_t)s * SLICE;
    #pragma unroll
    for (int nt = 0; nt < 4; ++nt) {
        const int b = nt * 16 + m;
        const float ymb = ymaxs[b];
        const float ipb = ipv[b];
        float mx = -INFINITY;
        #pragma unroll
        for (int v = 0; v < 8; ++v) {
            const int ig = wv * 16 + half * 8 + v;        // C layout: M = v + half*8
            const float val = __logf(c[nt][v] + 1e-12f) + ymb + ipb;
            outS[((size_t)ig * H + j) * B + b] = val;
            mx = fmaxf(mx, val);
        }
        atomicMax(&smax[b], encf(mx));
    }
    __syncthreads();
    if (tid < B) atomicMax(&bufCur[tid], smax[tid]);
}

// ---------------- final: y_final[b] over y_last with gamma softmax ----------
__global__ __launch_bounds__(256) void sohmm_final(const float* __restrict__ g,
                                                   const float* __restrict__ gstats,
                                                   float* __restrict__ out,
                                                   const unsigned* __restrict__ buf) {
    const int b   = blockIdx.x;
    const int tid = threadIdx.x;
    __shared__ float red[256];
    const float gmax = gstats[0];
    const float gsum = gstats[1];
    const float ymax = decf(buf[((S - 1) % 3) * B + b]);
    const float* ylast = out + (size_t)(S - 1) * SLICE;
    float acc = 0.f;
    for (int ij = tid; ij < H * H; ij += 256)
        acc += __expf(g[ij] - gmax + ylast[(size_t)ij * B + b] - ymax);
    red[tid] = acc; __syncthreads();
    for (int s = 128; s > 0; s >>= 1) {
        if (tid < s) red[tid] += red[tid + s];
        __syncthreads();
    }
    if (tid == 0)
        out[(size_t)S * SLICE + b] = __logf(red[0] / gsum + 1e-12f) + ymax;
}

// ---------------------------------------------------------------------------
extern "C" void kernel_launch(void* const* d_in, const int* in_sizes, int n_in,
                              void* d_out, int out_size, void* d_ws, size_t ws_size,
                              hipStream_t stream) {
    const int*   input_ids = (const int*)d_in[0];
    const float* alpha     = (const float*)d_in[1];
    const float* beta      = (const float*)d_in[2];
    const float* gamma     = (const float*)d_in[3];
    float*       out       = (float*)d_out;

    unsigned* buf    = (unsigned*)d_ws;                  // 3*64 encoded maxima
    float*    gstats = (float*)((char*)d_ws + 1024);     // {gmax, gsum}

    sohmm_gamma_stats<<<1, 256, 0, stream>>>(gamma, gstats);
    sohmm_init<<<1, 256, 0, stream>>>(buf);
    sohmm_y0<<<H, 256, 0, stream>>>(input_ids, beta, out, buf);
    for (int s = 1; s < S; ++s)
        sohmm_step<<<H, 256, 0, stream>>>(s, input_ids, alpha, beta, out, buf);
    sohmm_final<<<B, 256, 0, stream>>>(gamma, gstats, out, buf);
}
